// VGAE7_4277787427467
// MI455X (gfx1250) — matrix operand order repaired and checked by reading, not compiled, not run on hardware
//
#include <hip/hip_runtime.h>

// ---------------------------------------------------------------------------
// VGAE (3 GAT layers + MLP decoder) for MI455X / gfx1250.
// - bf16 v_wmma_f32_16x16x32_bf16 for all heavy matmuls
// - B-side operands kept K-major in global memory -> contiguous 16B LDS fills
// - async global->LDS copies (global_load_async_to_lds_b128 / s_wait_asynccnt)
// - flash-style softmax: per-row (max,sum) pass, then fused P@H
// ---------------------------------------------------------------------------

typedef unsigned short u16;
typedef unsigned int   u32;
typedef unsigned long long u64;
typedef __bf16 v16bf __attribute__((ext_vector_type(16)));
typedef float  v8f   __attribute__((ext_vector_type(8)));
typedef u32    u32x4 __attribute__((ext_vector_type(4)));

#define NN        4096
#define ALPHA     0.2f
#define NEG_INF   (-9.0e15f)

union U16frag { v16bf v; u16 u[16]; u32x4 q[2]; };

__device__ inline u16 f2bf(float f) {
    union { float f; u32 u; } c; c.f = f;
    u32 u = c.u + 0x7FFFu + ((c.u >> 16) & 1u);   // round-to-nearest-even
    return (u16)(u >> 16);
}
__device__ inline float bf2f(u16 h) {
    union { u32 u; float f; } c; c.u = ((u32)h) << 16;
    return c.f;
}

__device__ inline v8f wmma_bf16(v16bf a, v16bf b, v8f c) {
    return __builtin_amdgcn_wmma_f32_16x16x32_bf16(
        false, a, false, b, (short)0, c, false, false);
}

// Async 16B global->LDS copy (ASYNCcnt-tracked). LDS offset = low 32 bits of
// the generic shared pointer (ISA 10.2: LDS aperture addr[31:0]).
__device__ inline void cp_async16(u16* lds, const u16* gbl) {
    u32 loff = (u32)(size_t)lds;
    u64 ga   = (u64)(size_t)gbl;
    asm volatile("global_load_async_to_lds_b128 %0, %1, off"
                 :: "v"(loff), "v"(ga) : "memory");
}
__device__ inline void async_wait0() {
    asm volatile("s_wait_asynccnt 0x0" ::: "memory");
}

// A fragment (16x32 bf16, row-major tile in LDS, pitch 40 u16):
// lane<16: M=lane, elems 0..7 -> K 0..7, elems 8..15 -> K 16..23
// lane>=16: same M, K offset +8.  Two aligned 16B ds reads.
__device__ inline v16bf ldsA(const u16* base, int lane) {
    int l = lane & 15, lh = (lane >> 4) & 1;
    const u16* r = base + l * 40 + lh * 8;
    U16frag x;
    x.q[0] = *(const u32x4*)(r);
    x.q[1] = *(const u32x4*)(r + 16);
    return x.v;
}
// B fragment (32x16 bf16). LDS tile is K-transposed: Bt[n][k], pitch 40.
// lane<16: N=lane, elems 0..7 -> K 0..7, elems 8..15 -> K 16..23 ; lane>=16: K offset +8.  Two aligned 16B ds reads.
__device__ inline v16bf ldsB(const u16* base, int lane) {
    int l = lane & 15, lh = (lane >> 4) & 1;
    const u16* r = base + l * 40 + lh * 8;
    U16frag x;
    x.q[0] = *(const u32x4*)(r);
    x.q[1] = *(const u32x4*)(r + 16);
    return x.v;
}

// ---------------------------------------------------------------------------
// Elementwise / conversion kernels
// ---------------------------------------------------------------------------
__global__ void conv_bf16_kernel(const float* __restrict__ src,
                                 u16* __restrict__ dst, int n) {
    int i = blockIdx.x * 256 + threadIdx.x;
    if (i < n) dst[i] = f2bf(src[i]);
}

// Tiled transpose + convert + K-pad: src fp32 [Ks x Nc] row-major ->
// dst bf16 [Nc x Kp] K-major, dst[n][k] = (k < Ks) ? src[k][n] : 0
__global__ void convT_kernel(const float* __restrict__ src,
                             u16* __restrict__ dst, int Ks, int Kp, int Nc) {
    __shared__ float tile[32][33];
    int kb = blockIdx.x * 32, nb = blockIdx.y * 32;
    for (int r = threadIdx.y; r < 32; r += 8) {
        int k = kb + r, n = nb + threadIdx.x;
        tile[r][threadIdx.x] = (k < Ks && n < Nc) ? src[(size_t)k * Nc + n] : 0.f;
    }
    __syncthreads();
    for (int r = threadIdx.y; r < 32; r += 8) {
        int n = nb + r, k = kb + threadIdx.x;
        if (n < Nc && k < Kp) dst[(size_t)n * Kp + k] = f2bf(tile[threadIdx.x][r]);
    }
}

// hiddencat[:, 512:544]: Y into 512..527, zeros into 528..543
__global__ void hidc_tail_kernel(const float* __restrict__ Y,
                                 u16* __restrict__ hidc) {
    int i = blockIdx.x * 256 + threadIdx.x;     // over N*32
    if (i >= NN * 32) return;
    int r = i >> 5, c = i & 31;
    hidc[(size_t)r * 544 + 512 + c] = (c < 16) ? f2bf(Y[r * 16 + c]) : (u16)0;
}

// zcat[N,288]: cols 0..255 = noise*exp(logstd)+mean, 256..271 = Y, rest 0
__global__ void zcat_kernel(const float* __restrict__ noise,
                            const float* __restrict__ meanF,
                            const float* __restrict__ logF,
                            const float* __restrict__ Y,
                            u16* __restrict__ zc) {
    int i = blockIdx.x * 256 + threadIdx.x;     // over N*288
    if (i >= NN * 288) return;
    int r = i / 288, c = i - r * 288;
    u16 v;
    if (c < 256)      v = f2bf(noise[r * 256 + c] * __expf(logF[r * 256 + c]) +
                               meanF[r * 256 + c]);
    else if (c < 272) v = f2bf(Y[r * 16 + (c - 256)]);
    else              v = 0;
    zc[i] = v;
}

// ---------------------------------------------------------------------------
// f_src[i] = h[i,:].a[0:F], f_dst[i] = h[i,:].a[F:2F] from K-major HT [F x N]
// (one thread per node; fully coalesced: stride-N rows)
// ---------------------------------------------------------------------------
__global__ __launch_bounds__(256)
void fsd_kernel(const u16* __restrict__ HT, const float* __restrict__ a,
                int F, float* __restrict__ fs, float* __restrict__ fd) {
    int i = blockIdx.x * 256 + threadIdx.x;
    float sa = 0.f, sb = 0.f;
    for (int f = 0; f < F; ++f) {
        float h = bf2f(HT[(size_t)f * NN + i]);
        sa += h * a[f];
        sb += h * a[F + f];
    }
    fs[i] = sa; fd[i] = sb;
}

// ---------------------------------------------------------------------------
// Per-row online softmax stats over e_ij = adj? lrelu(fs_i+fd_j) : NEG_INF
// ---------------------------------------------------------------------------
__global__ __launch_bounds__(256)
void stats_kernel(const float* __restrict__ fs, const float* __restrict__ fd,
                  const unsigned char* __restrict__ adj,
                  float* __restrict__ mrow, float* __restrict__ srow) {
    __shared__ float rm[256], rs[256];
    int i = blockIdx.x;
    float fsi = fs[i];
    const unsigned char* arow = adj + (size_t)i * NN;
    float mx = -3.0e38f, sm = 0.f;
    for (int j = threadIdx.x; j < NN; j += 256) {
        float t = fsi + fd[j];
        float v = arow[j] ? (t > 0.f ? t : ALPHA * t) : NEG_INF;
        if (v > mx) { sm = sm * __expf(mx - v) + 1.f; mx = v; }
        else        { sm += __expf(v - mx); }
    }
    rm[threadIdx.x] = mx; rs[threadIdx.x] = sm;
    __syncthreads();
    for (int off = 128; off; off >>= 1) {
        if (threadIdx.x < off) {
            float m1 = rm[threadIdx.x], s1 = rs[threadIdx.x];
            float m2 = rm[threadIdx.x + off], s2 = rs[threadIdx.x + off];
            float mn = fmaxf(m1, m2);
            rm[threadIdx.x] = mn;
            rs[threadIdx.x] = s1 * __expf(m1 - mn) + s2 * __expf(m2 - mn);
        }
        __syncthreads();
    }
    if (threadIdx.x == 0) { mrow[i] = rm[0]; srow[i] = rs[0]; }
}

// ---------------------------------------------------------------------------
// bf16 WMMA GEMM: C[M,N] = A[M,K] (row-major) * BT[N,K] (K-major) (+bias,act)
// 256 thr = 8 waves; macro-tile 256(M) x 64(N); each wave: 2 M-frags x 4 N-frags
// = 8 wmma per K-step (B fragments reused across both M-frags).
// EPI: 0 raw->bf16 K-major [Nn x ldo]  3 bias+relu->bf16 row-major
//      4 bias+sigmoid->fp32 row-major
// ---------------------------------------------------------------------------
template<int EPI>
__global__ __launch_bounds__(256)
void gemm_kernel(const u16* __restrict__ A, const u16* __restrict__ BT,
                 const float* __restrict__ bias,
                 u16* __restrict__ outB, float* __restrict__ outF,
                 int M, int Nn, int K, int ldo) {
    __shared__ __align__(16) u16 As[256 * 40];
    __shared__ __align__(16) u16 Bt[64 * 40];
    const int tid = threadIdx.x;
    const int wave = tid >> 5, lane = tid & 31;
    const int l = lane & 15, lh = (lane >> 4) & 1;
    const int m0 = blockIdx.y * 256;
    const int n0 = blockIdx.x * 64;
    v8f acc[2][4] = {};

    for (int k0 = 0; k0 < K; k0 += 32) {
        if (k0 + 32 < K) {
            __builtin_prefetch(A + (size_t)(m0 + (tid >> 1)) * K + k0 + 32, 0, 1);
            __builtin_prefetch(BT + (size_t)(n0 + (tid >> 3)) * K + k0 + 32, 0, 1);
        }
        // A tile 256x32 (async, 4 x 16B per thread)
        #pragma unroll
        for (int i = 0; i < 4; ++i) {
            int c = tid + i * 256;               // 1024 16B chunks
            int row = c >> 2, kg = (c & 3) * 8;
            cp_async16(As + row * 40 + kg,
                       A + (size_t)(m0 + row) * K + k0 + kg);
        }
        // B tile 64x32, already K-major in memory (1 x 16B per thread)
        {
            int n = tid >> 2, kg = (tid & 3) * 8;
            cp_async16(Bt + n * 40 + kg,
                       BT + (size_t)(n0 + n) * K + k0 + kg);
        }
        async_wait0();
        __syncthreads();
        v16bf a0 = ldsA(As + (wave * 32) * 40, lane);
        v16bf a1 = ldsA(As + (wave * 32 + 16) * 40, lane);
        #pragma unroll
        for (int ns = 0; ns < 4; ++ns) {
            v16bf b = ldsB(Bt + (ns * 16) * 40, lane);
            acc[0][ns] = wmma_bf16(a0, b, acc[0][ns]);
            acc[1][ns] = wmma_bf16(a1, b, acc[1][ns]);
        }
        __syncthreads();
    }
    #pragma unroll
    for (int mf = 0; mf < 2; ++mf) {
        #pragma unroll
        for (int ns = 0; ns < 4; ++ns) {
            int col = n0 + ns * 16 + l;
            float bv = (EPI >= 3) ? bias[col] : 0.f;
            #pragma unroll
            for (int r = 0; r < 8; ++r) {
                int row = m0 + wave * 32 + mf * 16 + r + lh * 8;
                float v = acc[mf][ns][r];
                if (EPI == 3) {
                    v += bv; v = v > 0.f ? v : 0.f;
                    outB[(size_t)row * ldo + col] = f2bf(v);
                } else if (EPI == 4) {
                    v = 1.f / (1.f + __expf(-(v + bv)));
                    outF[(size_t)row * ldo + col] = v;
                } else {
                    // K-major activation store: out[col][row], ldo = M
                    outB[(size_t)col * ldo + row] = f2bf(v);
                }
            }
        }
    }
}

// ---------------------------------------------------------------------------
// Fused attention aggregation: out[i,f] = elu( sum_j P_ij * H[j,f] )
// P_ij = exp((adj? lrelu(fs_i+fd_j):NEG_INF) - m_i) / s_i, built once per
// 32-row block as bf16 tiles in LDS; H consumed K-major (HT[f][j]) via async
// 16B copies. Each wave: 2 M-frags x NS N-frags.
// EPI: 1 elu->bf16 row-major (hiddencat, pitched)  2 elu->fp32 row-major
// ---------------------------------------------------------------------------
template<int EPI, int F>
__global__ __launch_bounds__(256)
void attn_kernel(const float* __restrict__ fs, const float* __restrict__ fd,
                 const float* __restrict__ mrow, const float* __restrict__ srow,
                 const unsigned char* __restrict__ adj,
                 const u16* __restrict__ HT,
                 u16* __restrict__ outB, float* __restrict__ outF, int ldo) {
    extern __shared__ __align__(16) u16 smem[];
    u16* Pt = smem;                 // 32 x 40
    u16* Ht = smem + 32 * 40;       // F  x 40 (K-transposed H tile)
    const int tid = threadIdx.x;
    const int wave = tid >> 5, lane = tid & 31;
    const int l = lane & 15, lh = (lane >> 4) & 1;
    const int r0 = blockIdx.x * 32;
    constexpr int FW = F / 8;       // columns per wave
    constexpr int NS = FW / 16;     // N-frags per wave
    v8f acc[2][NS] = {};

    // Hoist row stats (row index fixed per thread slot)
    float frr[4], inv_s[4], mr[4];
    const unsigned char* arow[4];
    #pragma unroll
    for (int i = 0; i < 4; ++i) {
        int rr = (tid + i * 256) >> 5;          // 0..31
        frr[i] = fs[r0 + rr];
        mr[i] = mrow[r0 + rr];
        inv_s[i] = 1.f / srow[r0 + rr];
        arow[i] = adj + (size_t)(r0 + rr) * NN;
    }

    for (int k0 = 0; k0 < NN; k0 += 32) {
        // P tile 32x32 (each thread: 4 values)
        #pragma unroll
        for (int i = 0; i < 4; ++i) {
            int idx = tid + i * 256;
            int rr = idx >> 5, kk = idx & 31;
            float t = frr[i] + fd[k0 + kk];
            float v = arow[i][k0 + kk] ? (t > 0.f ? t : ALPHA * t) : NEG_INF;
            Pt[rr * 40 + kk] = f2bf(__expf(v - mr[i]) * inv_s[i]);
        }
        // H tile F x 32 from K-major HT (async contiguous 16B copies)
        #pragma unroll
        for (int c = 0; c < F / 64; ++c) {
            int ch = tid + c * 256;             // F*4 chunks
            int n = ch >> 2, kg = (ch & 3) * 8;
            cp_async16(Ht + n * 40 + kg, HT + (size_t)n * NN + k0 + kg);
        }
        async_wait0();
        __syncthreads();
        v16bf a0 = ldsA(Pt, lane);
        v16bf a1 = ldsA(Pt + 16 * 40, lane);
        #pragma unroll
        for (int ns = 0; ns < NS; ++ns) {
            v16bf b = ldsB(Ht + (wave * FW + ns * 16) * 40, lane);
            acc[0][ns] = wmma_bf16(a0, b, acc[0][ns]);
            acc[1][ns] = wmma_bf16(a1, b, acc[1][ns]);
        }
        __syncthreads();
    }
    #pragma unroll
    for (int mf = 0; mf < 2; ++mf) {
        #pragma unroll
        for (int ns = 0; ns < NS; ++ns) {
            int col = wave * FW + ns * 16 + l;
            #pragma unroll
            for (int r = 0; r < 8; ++r) {
                int row = r0 + mf * 16 + r + lh * 8;
                float v = acc[mf][ns][r];
                float e = v > 0.f ? v : (__expf(v) - 1.f);   // ELU
                if (EPI == 1) outB[(size_t)row * ldo + col] = f2bf(e);
                else          outF[(size_t)row * F + col] = e;
            }
        }
    }
}

// ---------------------------------------------------------------------------
// Host launcher
// ---------------------------------------------------------------------------
extern "C" void kernel_launch(void* const* d_in, const int* in_sizes, int n_in,
                              void* d_out, int out_size, void* d_ws, size_t ws_size,
                              hipStream_t stream) {
    const float* X     = (const float*)d_in[0];   // [4096,1024]
    const float* Y     = (const float*)d_in[1];   // [4096,16]
    const float* noise = (const float*)d_in[2];   // [4096,256]
    const unsigned char* adj = (const unsigned char*)d_in[3]; // bool [4096,4096]
    const float* W0    = (const float*)d_in[4];   // [1024,512]
    const float* a0    = (const float*)d_in[5];   // [1024]
    const float* W1    = (const float*)d_in[6];   // [528,256]
    const float* a1    = (const float*)d_in[7];   // [512]
    const float* W2    = (const float*)d_in[8];   // [528,256]
    const float* a2    = (const float*)d_in[9];   // [512]
    const float* fc1w  = (const float*)d_in[10];  // [272,512]
    const float* fc1b  = (const float*)d_in[11];  // [512]
    const float* fc2w  = (const float*)d_in[12];  // [512,4096]
    const float* fc2b  = (const float*)d_in[13];  // [4096]
    float* out = (float*)d_out;                   // [4096,4096]

    char* ws = (char*)d_ws;
    size_t off = 0;
    auto carve = [&](size_t bytes) -> void* {
        off = (off + 255) & ~(size_t)255;
        void* p = ws + off;
        off += bytes;
        return p;
    };
    u16* Xb     = (u16*)carve((size_t)NN * 1024 * 2);  // row-major [N,1024]
    u16* W0bT   = (u16*)carve((size_t)512 * 1024 * 2); // K-major  [512,1024]
    u16* h0bT   = (u16*)carve((size_t)512 * NN * 2);   // K-major  [512,N]
    u16* hidc   = (u16*)carve((size_t)NN * 544 * 2);   // row-major [N,544]
    u16* W1bT   = (u16*)carve((size_t)256 * 544 * 2);  // K-major
    u16* W2bT   = (u16*)carve((size_t)256 * 544 * 2);
    u16* h1bT   = (u16*)carve((size_t)256 * NN * 2);   // K-major
    u16* h2bT   = (u16*)carve((size_t)256 * NN * 2);
    u16* fc1wbT = (u16*)carve((size_t)512 * 288 * 2);  // K-major
    u16* fc2wbT = (u16*)carve((size_t)4096 * 512 * 2); // K-major
    u16* zcb    = (u16*)carve((size_t)NN * 288 * 2);   // row-major
    u16* zhb    = (u16*)carve((size_t)NN * 512 * 2);   // row-major
    float* meanF = (float*)carve((size_t)NN * 256 * 4);
    float* logF  = (float*)carve((size_t)NN * 256 * 4);
    float* fs0 = (float*)carve(NN * 4); float* fd0 = (float*)carve(NN * 4);
    float* m0  = (float*)carve(NN * 4); float* s0  = (float*)carve(NN * 4);
    float* fs1 = (float*)carve(NN * 4); float* fd1 = (float*)carve(NN * 4);
    float* m1  = (float*)carve(NN * 4); float* s1  = (float*)carve(NN * 4);
    float* fs2 = (float*)carve(NN * 4); float* fd2 = (float*)carve(NN * 4);
    float* m2  = (float*)carve(NN * 4); float* s2  = (float*)carve(NN * 4);

    auto cdiv = [](int a, int b) { return (a + b - 1) / b; };
    dim3 tb(32, 8);

    // --- conversions (weights -> K-major bf16, with K zero-padding) ---
    conv_bf16_kernel<<<cdiv(NN * 1024, 256), 256, 0, stream>>>(X, Xb, NN * 1024);
    convT_kernel<<<dim3(cdiv(1024, 32), cdiv(512, 32)), tb, 0, stream>>>(
        W0, W0bT, 1024, 1024, 512);
    convT_kernel<<<dim3(cdiv(544, 32), cdiv(256, 32)), tb, 0, stream>>>(
        W1, W1bT, 528, 544, 256);
    convT_kernel<<<dim3(cdiv(544, 32), cdiv(256, 32)), tb, 0, stream>>>(
        W2, W2bT, 528, 544, 256);
    convT_kernel<<<dim3(cdiv(288, 32), cdiv(512, 32)), tb, 0, stream>>>(
        fc1w, fc1wbT, 272, 288, 512);
    convT_kernel<<<dim3(cdiv(512, 32), cdiv(4096, 32)), tb, 0, stream>>>(
        fc2w, fc2wbT, 512, 512, 4096);
    hidc_tail_kernel<<<cdiv(NN * 32, 256), 256, 0, stream>>>(Y, hidc);

    // --- GAT layer 0 ---
    gemm_kernel<0><<<dim3(512 / 64, NN / 256), 256, 0, stream>>>(
        Xb, W0bT, nullptr, h0bT, nullptr, NN, 512, 1024, NN);
    fsd_kernel<<<NN / 256, 256, 0, stream>>>(h0bT, a0, 512, fs0, fd0);
    stats_kernel<<<NN, 256, 0, stream>>>(fs0, fd0, adj, m0, s0);
    attn_kernel<1, 512><<<NN / 32, 256, (32 + 512) * 40 * 2, stream>>>(
        fs0, fd0, m0, s0, adj, h0bT, hidc, nullptr, 544);

    // --- GAT layers 1 & 2 (mean / logstd) ---
    gemm_kernel<0><<<dim3(256 / 64, NN / 256), 256, 0, stream>>>(
        hidc, W1bT, nullptr, h1bT, nullptr, NN, 256, 544, NN);
    gemm_kernel<0><<<dim3(256 / 64, NN / 256), 256, 0, stream>>>(
        hidc, W2bT, nullptr, h2bT, nullptr, NN, 256, 544, NN);
    fsd_kernel<<<NN / 256, 256, 0, stream>>>(h1bT, a1, 256, fs1, fd1);
    fsd_kernel<<<NN / 256, 256, 0, stream>>>(h2bT, a2, 256, fs2, fd2);
    stats_kernel<<<NN, 256, 0, stream>>>(fs1, fd1, adj, m1, s1);
    stats_kernel<<<NN, 256, 0, stream>>>(fs2, fd2, adj, m2, s2);
    attn_kernel<2, 256><<<NN / 32, 256, (32 + 256) * 40 * 2, stream>>>(
        fs1, fd1, m1, s1, adj, h1bT, nullptr, meanF, 0);
    attn_kernel<2, 256><<<NN / 32, 256, (32 + 256) * 40 * 2, stream>>>(
        fs2, fd2, m2, s2, adj, h2bT, nullptr, logF, 0);

    // --- reparameterize + decoder ---
    zcat_kernel<<<cdiv(NN * 288, 256), 256, 0, stream>>>(noise, meanF, logF, Y, zcb);
    gemm_kernel<3><<<dim3(512 / 64, NN / 256), 256, 0, stream>>>(
        zcb, fc1wbT, fc1b, zhb, nullptr, NN, 512, 288, 512);
    gemm_kernel<4><<<dim3(4096 / 64, NN / 256), 256, 0, stream>>>(
        zhb, fc2wbT, fc2b, nullptr, out, NN, 4096, 512, 4096);
}
